// WSAD_67817533604438
// MI455X (gfx1250) — compile-verified, matrix operand order checked by reading, not run on hardware
//
#include <hip/hip_runtime.h>
#include <hip/hip_bf16.h>
#include <math.h>

// ---------------------------------------------------------------------------
// CDNA5 / gfx1250: all contractions via v_wmma_f32_16x16x32_f16 (f32 accum),
// f16 operand staging through LDS using global_load_async_to_lds_b128.
// ---------------------------------------------------------------------------

typedef __attribute__((ext_vector_type(8)))  _Float16 v8h;
typedef __attribute__((ext_vector_type(16))) _Float16 v16h;
typedef __attribute__((ext_vector_type(8)))  float    v8f;
typedef __attribute__((ext_vector_type(4)))  int      v4i;

#define TILE_M 64
#define TILE_N 64
#define TILE_K 32
#define LDSROW 40                     // halves per LDS row: 32 data + 8 pad (80B, 16B-aligned)
#define GEMM_SMEM (2 * 64 * LDSROW * 2)

#if defined(__HIP_DEVICE_COMPILE__) && __has_builtin(__builtin_amdgcn_global_load_async_to_lds_b128) && __has_builtin(__builtin_amdgcn_s_wait_asynccnt)
#define USE_ASYNC 1
#else
#define USE_ASYNC 0
#endif

__device__ __forceinline__ void copy16B_to_lds(const _Float16* src, _Float16* dst)
{
#if USE_ASYNC
    __builtin_amdgcn_global_load_async_to_lds_b128(
        (__attribute__((address_space(1))) v4i*)(src),
        (__attribute__((address_space(3))) v4i*)(dst), 0, 0);
#else
    *(v8h*)dst = *(const v8h*)src;
#endif
}

// act: 0=none 1=relu 2=gelu(exact) 3=sigmoid
// C = act(alpha * A@B + bias) [+ residual];  A,B are f16; C written f32 and/or f16.
// B element (k,n) at B[k*ldbk + n*ldbn]; exactly one of ldbk/ldbn is 1.
// Requires M%64==0, N%64==0, K%32==0 (caller pads).
__global__ __launch_bounds__(128)
void gemm_wmma_f16(const _Float16* __restrict__ A, const _Float16* __restrict__ B,
                   const float* __restrict__ bias, const float* __restrict__ residual,
                   float* __restrict__ C32, _Float16* __restrict__ C16,
                   int M, int N, int K,
                   long lda, long ldbk, long ldbn, long ldc,
                   long sAb, long sAh, long sBb, long sBh, long sCb, long sCh, int H,
                   float alpha, int act)
{
    extern __shared__ _Float16 smem[];
    _Float16* As = smem;               // [64][LDSROW]
    _Float16* Bs = smem + 64 * LDSROW; // [64][LDSROW]  (stored [n][k])

    const int z  = blockIdx.z;
    const int zb = z / H, zh = z % H;
    A += (long)zb * sAb + (long)zh * sAh;
    B += (long)zb * sBb + (long)zh * sBh;
    const long coff = (long)zb * sCb + (long)zh * sCh;
    if (C32) C32 += coff;
    if (C16) C16 += coff;
    const float* res = residual ? residual + coff : nullptr;

    const int bM = blockIdx.y * TILE_M;
    const int bN = blockIdx.x * TILE_N;

    const int tid  = threadIdx.x;          // 128 = 4 wave32 waves
    const int wave = tid >> 5;
    const int lane = tid & 31;
    const int wm   = (wave >> 1) * 32;
    const int wn   = (wave & 1) * 32;
    const int row  = lane & 15;
    const int ksel = (lane >> 4) * 8;      // ISA 16-bit fragment K grouping

    // ---- per-thread staging addresses (2 x 16B chunks per operand) ----
    const int ar = tid >> 2, aq = tid & 3;
    const _Float16* srcA0 = A + (long)(bM + ar) * lda + aq * 8;
    const _Float16* srcA1 = A + (long)(bM + ar + 32) * lda + aq * 8;
    _Float16* dstA0 = As + ar * LDSROW + aq * 8;
    _Float16* dstA1 = As + (ar + 32) * LDSROW + aq * 8;

    const bool bfast = (ldbk == 1);
    const _Float16* srcB0 = nullptr;
    const _Float16* srcB1 = nullptr;
    _Float16* dstB0 = nullptr;
    _Float16* dstB1 = nullptr;
    int bk = 0, bo = 0;
    if (bfast) {
        srcB0 = B + (long)(bN + ar) * ldbn + aq * 8;
        srcB1 = B + (long)(bN + ar + 32) * ldbn + aq * 8;
        dstB0 = Bs + ar * LDSROW + aq * 8;
        dstB1 = Bs + (ar + 32) * LDSROW + aq * 8;
    } else {
        bk = tid >> 3; bo = tid & 7;
        srcB0 = B + (long)bk * ldbk + bN + bo * 8;
        srcB1 = B + (long)(bk + 16) * ldbk + bN + bo * 8;
    }
    const long bstep = (long)TILE_K * ldbk;

    v8f acc[2][2] = {};

    for (int k0 = 0; k0 < K; k0 += TILE_K) {
        // ---- A tile (branchless, pointer-incremented) ----
        copy16B_to_lds(srcA0, dstA0);
        copy16B_to_lds(srcA1, dstA1);
        srcA0 += TILE_K; srcA1 += TILE_K;
        // ---- B tile into Bs[n][k] ----
        if (bfast) {
            copy16B_to_lds(srcB0, dstB0);
            copy16B_to_lds(srcB1, dstB1);
            srcB0 += TILE_K; srcB1 += TILE_K;
        } else {
            v8h t0 = *(const v8h*)srcB0;
            v8h t1 = *(const v8h*)srcB1;
            #pragma unroll
            for (int j = 0; j < 8; ++j) {
                Bs[(bo * 8 + j) * LDSROW + bk]      = t0[j];
                Bs[(bo * 8 + j) * LDSROW + bk + 16] = t1[j];
            }
            srcB0 += bstep; srcB1 += bstep;
        }
#if USE_ASYNC
        __builtin_amdgcn_s_wait_asynccnt(0);
#endif
        __syncthreads();

        // ---- fragments: two ds_load_b128 each ----
        v8h alo[2], ahi[2], blo[2], bhi[2];
        #pragma unroll
        for (int i = 0; i < 2; ++i) {
            const _Float16* pa = As + (wm + i * 16 + row) * LDSROW + ksel;
            alo[i] = *(const v8h*)pa;
            ahi[i] = *(const v8h*)(pa + 16);
            const _Float16* pb = Bs + (wn + i * 16 + row) * LDSROW + ksel;
            blo[i] = *(const v8h*)pb;
            bhi[i] = *(const v8h*)(pb + 16);
        }
        #pragma unroll
        for (int i = 0; i < 2; ++i) {
            v16h af = __builtin_shufflevector(alo[i], ahi[i],
                        0,1,2,3,4,5,6,7,8,9,10,11,12,13,14,15);
            #pragma unroll
            for (int j = 0; j < 2; ++j) {
                v16h bf = __builtin_shufflevector(blo[j], bhi[j],
                            0,1,2,3,4,5,6,7,8,9,10,11,12,13,14,15);
                acc[i][j] = __builtin_amdgcn_wmma_f32_16x16x32_f16(
                    false, af, false, bf, (short)0, acc[i][j], false, false);
            }
        }
        __syncthreads();
    }

    // ---- epilogue ----
    const int rhi = (lane >> 4) * 8;       // C/D layout: M = r + 8*(lane>=16)
    #pragma unroll
    for (int i = 0; i < 2; ++i) {
        #pragma unroll
        for (int j = 0; j < 2; ++j) {
            int gn = bN + wn + j * 16 + row;
            float bv = bias ? bias[gn] : 0.0f;
            #pragma unroll
            for (int r = 0; r < 8; ++r) {
                int gm = bM + wm + i * 16 + rhi + r;
                float v = acc[i][j][r] * alpha + bv;
                if      (act == 1) v = fmaxf(v, 0.0f);
                else if (act == 2) v = 0.5f * v * (1.0f + erff(v * 0.70710678118f));
                else if (act == 3) v = 1.0f / (1.0f + __expf(-v));
                long off = (long)gm * ldc + gn;
                if (res) v += res[off];
                if (C32) C32[off] = v;
                if (C16) C16[off] = (_Float16)v;
            }
        }
    }
}

// ---------------------------------------------------------------------------
// Operand preparation / support kernels
// ---------------------------------------------------------------------------

__global__ void cvt_f16(const float* __restrict__ in, _Float16* __restrict__ out, long n)
{
    long i = (long)blockIdx.x * blockDim.x + threadIdx.x;
    if (i < n) out[i] = (_Float16)in[i];
}

// out[n*K + k] = in[k*N + n]  (weight transpose to [n][k], f32 -> f16)
__global__ void cvtT_f16(const float* __restrict__ in, _Float16* __restrict__ out, int K, int N)
{
    long i = (long)blockIdx.x * blockDim.x + threadIdx.x;
    if (i >= (long)K * N) return;
    int n = (int)(i % N), k = (int)(i / N);
    out[(long)n * K + k] = (_Float16)in[i];
}

// mem (60,512) -> padded [64][512] (row-major) and transposed [512][64]
__global__ void pad_mem(const float* __restrict__ mem, _Float16* __restrict__ outA,
                        _Float16* __restrict__ outT)
{
    long i = (long)blockIdx.x * blockDim.x + threadIdx.x;   // 64*512
    if (i >= 64L * 512) return;
    int r = (int)(i >> 9), d = (int)(i & 511);
    float v = (r < 60) ? mem[r * 512 + d] : 0.0f;
    outA[i] = (_Float16)v;
    outT[(long)d * 64 + r] = (_Float16)v;
}

// x:(64,256,1024) -> Xcol16:(16384, 3072), Kidx = i*3 + kk (matches conv_w flat)
__global__ void im2col_f16(const float* __restrict__ x, _Float16* __restrict__ xc, long total)
{
    long idx = (long)blockIdx.x * blockDim.x + threadIdx.x;
    if (idx >= total) return;
    long m = idx / 3072;
    int  c = (int)(idx % 3072);
    int  i = c / 3, kk = c % 3;
    int  t = (int)(m & 255);
    long b = m >> 8;
    int  ts = t + kk - 1;
    float v = 0.0f;
    if (ts >= 0 && ts < 256) v = x[(b * 256 + ts) * 1024 + i];
    xc[idx] = (_Float16)v;
}

__global__ __launch_bounds__(256)
void layernorm512(const float* __restrict__ x, const float* __restrict__ g,
                  const float* __restrict__ b, _Float16* __restrict__ o)
{
    __shared__ float red[256];
    long row = blockIdx.x;
    const float* p = x + row * 512;
    int t = threadIdx.x;
    float v0 = p[t], v1 = p[t + 256];
    red[t] = v0 + v1; __syncthreads();
    for (int st = 128; st; st >>= 1) { if (t < st) red[t] += red[t + st]; __syncthreads(); }
    float mean = red[0] * (1.0f / 512.0f); __syncthreads();
    float d0 = v0 - mean, d1 = v1 - mean;
    red[t] = d0 * d0 + d1 * d1; __syncthreads();
    for (int st = 128; st; st >>= 1) { if (t < st) red[t] += red[t + st]; __syncthreads(); }
    float rstd = rsqrtf(red[0] * (1.0f / 512.0f) + 1e-5f);
    o[row * 512 + t]       = (_Float16)(d0 * rstd * g[t]       + b[t]);
    o[row * 512 + t + 256] = (_Float16)(d1 * rstd * g[t + 256] + b[t + 256]);
}

__global__ __launch_bounds__(256)
void softmax256(const float* __restrict__ s, _Float16* __restrict__ p16)
{
    __shared__ float red[256];
    long row = blockIdx.x;
    const float* p = s + row * 256;
    int t = threadIdx.x;
    float v = p[t];
    red[t] = v; __syncthreads();
    for (int st = 128; st; st >>= 1) { if (t < st) red[t] = fmaxf(red[t], red[t + st]); __syncthreads(); }
    float mx = red[0]; __syncthreads();
    float e = __expf(v - mx);
    red[t] = e; __syncthreads();
    for (int st = 128; st; st >>= 1) { if (t < st) red[t] += red[t + st]; __syncthreads(); }
    p16[row * 256 + t] = (_Float16)(e / red[0]);
}

// per-row top-4 mean over cols (row stride ld)
__global__ void top4mean(const float* __restrict__ att, float* __restrict__ out,
                         int rows, int cols, int ld)
{
    int r = blockIdx.x * blockDim.x + threadIdx.x;
    if (r >= rows) return;
    const float* p = att + (long)r * ld;
    float t0 = -1e30f, t1 = -1e30f, t2 = -1e30f, t3 = -1e30f;
    for (int j = 0; j < cols; ++j) {
        float v = p[j];
        if      (v > t0) { t3 = t2; t2 = t1; t1 = t0; t0 = v; }
        else if (v > t1) { t3 = t2; t2 = t1; t1 = v; }
        else if (v > t2) { t3 = t2; t2 = v; }
        else if (v > t3) { t3 = v; }
    }
    out[r] = (t0 + t1 + t2 + t3) * 0.25f;
}

__global__ void topk_idx(const float* __restrict__ att, int* __restrict__ idx, int B, int T, int k)
{
    int b = blockIdx.x * blockDim.x + threadIdx.x;
    if (b >= B) return;
    const float* p = att + (long)b * T;
    bool used[256];
    for (int j = 0; j < T; ++j) used[j] = false;
    for (int s = 0; s < k; ++s) {
        float best = -1e30f; int bi = 0;
        for (int j = 0; j < T; ++j)
            if (!used[j] && p[j] > best) { best = p[j]; bi = j; }
        used[bi] = true;
        idx[(long)b * k + s] = bi;
    }
}

__global__ void gather_mean(const float* __restrict__ feats, const int* __restrict__ idx,
                            float* __restrict__ out, int T, int Dm, int k)
{
    int b = blockIdx.x;
    int d = blockIdx.y * blockDim.x + threadIdx.x;
    if (d >= Dm) return;
    float s = 0.0f;
    for (int j = 0; j < k; ++j)
        s += feats[((long)b * T + idx[b * k + j]) * Dm + d];
    out[(long)b * Dm + d] = s * (1.0f / k);
}

__global__ void vae_combine(const float* __restrict__ mu, const float* __restrict__ var,
                            const float* __restrict__ eps, float* __restrict__ out, long n)
{
    long i = (long)blockIdx.x * blockDim.x + threadIdx.x;
    if (i >= n) return;
    out[i] = mu[i] + eps[i] * sqrtf(__expf(var[i]));
}

__global__ __launch_bounds__(256)
void kl_partial(const float* __restrict__ mu, const float* __restrict__ var,
                float* __restrict__ part, long n)
{
    __shared__ float red[256];
    long i0 = ((long)blockIdx.x * 256 + threadIdx.x) * 4;
    float s = 0.0f;
    for (int j = 0; j < 4; ++j) {
        long i = i0 + j;
        if (i < n) { float m = mu[i], v = var[i]; s += 1.0f + v - m * m - __expf(v); }
    }
    red[threadIdx.x] = s; __syncthreads();
    for (int st = 128; st; st >>= 1) { if (threadIdx.x < st) red[threadIdx.x] += red[threadIdx.x + st]; __syncthreads(); }
    if (threadIdx.x == 0) part[blockIdx.x] = red[0];
}

__global__ __launch_bounds__(256)
void tripcos_kernel(const float* __restrict__ anchor, const float* __restrict__ positive,
                    const float* __restrict__ negative, float* __restrict__ trip_b,
                    float* __restrict__ cos_b)
{
    __shared__ float red[256];
    int b = blockIdx.x, t = threadIdx.x;
    const float* a = anchor   + (long)b * 512;
    const float* p = positive + (long)b * 512;
    const float* g = negative + (long)b * 512;
    float a0 = a[t], a1 = a[t + 256];
    float p0 = p[t], p1 = p[t + 256];
    float g0 = g[t], g1 = g[t + 256];

    auto reduce = [&](float v) -> float {
        red[t] = v; __syncthreads();
        for (int st = 128; st; st >>= 1) { if (t < st) red[t] += red[t + st]; __syncthreads(); }
        float r = red[0]; __syncthreads();
        return r;
    };

    float na2 = reduce(a0 * a0 + a1 * a1);
    float np2 = reduce(p0 * p0 + p1 * p1);
    float ng2 = reduce(g0 * g0 + g1 * g1);
    float rna = rsqrtf(na2), rnp = rsqrtf(np2), rng = rsqrtf(ng2);
    const float eps = 1e-6f;
    float d0 = a0 * rna - p0 * rnp + eps, d1 = a1 * rna - p1 * rnp + eps;
    float dp2 = reduce(d0 * d0 + d1 * d1);
    float e0 = a0 * rna - g0 * rng + eps, e1 = a1 * rna - g1 * rng + eps;
    float dn2 = reduce(e0 * e0 + e1 * e1);
    float dot = reduce(a0 * g0 + a1 * g1);
    if (t == 0) {
        trip_b[b] = fmaxf(sqrtf(dp2) - sqrtf(dn2) + 1.0f, 0.0f);
        float na = fmaxf(sqrtf(na2), eps), ngn = fmaxf(sqrtf(ng2), eps);
        cos_b[b] = 1.0f - dot / (na * ngn);
    }
}

__global__ __launch_bounds__(256)
void dist_kernel(const float* __restrict__ anc_new, const float* __restrict__ neg_new,
                 float* __restrict__ dist_b)
{
    __shared__ float red[256];
    int b = blockIdx.x, t = threadIdx.x;
    float a0 = anc_new[(long)b * 512 + t], a1 = anc_new[(long)b * 512 + t + 256];
    float g0 = neg_new[(long)b * 512 + t], g1 = neg_new[(long)b * 512 + t + 256];
    red[t] = a0 * a0 + a1 * a1; __syncthreads();
    for (int st = 128; st; st >>= 1) { if (t < st) red[t] += red[t + st]; __syncthreads(); }
    float na = sqrtf(red[0]); __syncthreads();
    red[t] = g0 * g0 + g1 * g1; __syncthreads();
    for (int st = 128; st; st >>= 1) { if (t < st) red[t] += red[t + st]; __syncthreads(); }
    float ng = sqrtf(red[0]);
    if (t == 0) dist_b[b] = fmaxf(100.0f - ng + na, 0.0f);
}

__global__ __launch_bounds__(256)
void finalize_scalars(const float* __restrict__ kl_part, int nkp,
                      const float* __restrict__ trip_b, const float* __restrict__ cos_b,
                      const float* __restrict__ dist_b, float* __restrict__ out, long cos_off)
{
    __shared__ float red[256];
    int t = threadIdx.x;
    float s = 0.0f;
    for (int i = t; i < nkp; i += 256) s += kl_part[i];
    red[t] = s; __syncthreads();
    for (int st = 128; st; st >>= 1) { if (t < st) red[t] += red[t + st]; __syncthreads(); }
    if (t == 0) {
        out[1] = -0.5f * red[0] / (32.0f * 512.0f);
        float tr = 0, co = 0, di = 0;
        for (int b = 0; b < 32; ++b) { tr += trip_b[b]; co += cos_b[b]; di += dist_b[b]; }
        out[0]       = tr * (1.0f / 32.0f);
        out[2]       = di * (1.0f / 32.0f);
        out[cos_off] = co * (1.0f / 32.0f);
    }
}

__global__ void assemble_out(const float* __restrict__ x_emb,
                             const float* __restrict__ N_aug_new, const float* __restrict__ A_Naug_m,
                             const float* __restrict__ A_aug_new, const float* __restrict__ N_Aaug_m,
                             float* __restrict__ xout, float* __restrict__ vfeat)
{
    long i = (long)blockIdx.x * blockDim.x + threadIdx.x;   // 16384*512
    if (i >= 16384L * 512) return;
    long m = i >> 9; int d = (int)(i & 511);
    long b = m >> 8; int t = (int)(m & 255);
    float e = x_emb[i];
    float aug;
    if (m < 8192) aug = N_aug_new[i] + A_Naug_m[i];
    else { long j = i - 8192L * 512; aug = A_aug_new[j] + N_Aaug_m[j]; }
    xout[m * 1024 + d]       = e;
    xout[m * 1024 + 512 + d] = aug;
    vfeat[(b * 512 + d) * 256 + t] = e;
}

// ---------------------------------------------------------------------------
// Host orchestration
// ---------------------------------------------------------------------------

static inline void launch_gemm(hipStream_t s, const _Float16* A, const _Float16* B,
                               const float* bias, const float* res,
                               float* C32, _Float16* C16, int M, int N, int K,
                               long lda, long ldbk, long ldbn, long ldc,
                               long sAb, long sAh, long sBb, long sBh, long sCb, long sCh,
                               int H, int Z, float alpha, int act)
{
    dim3 g(N / TILE_N, M / TILE_M, Z);
    gemm_wmma_f16<<<g, 128, GEMM_SMEM, s>>>(A, B, bias, res, C32, C16, M, N, K,
                                            lda, ldbk, ldbn, ldc,
                                            sAb, sAh, sBb, sBh, sCb, sCh, H, alpha, act);
}

extern "C" void kernel_launch(void* const* d_in, const int* in_sizes, int n_in,
                              void* d_out, int out_size, void* d_ws, size_t ws_size,
                              hipStream_t stream)
{
    (void)in_sizes; (void)n_in; (void)out_size; (void)ws_size;

    const float* x      = (const float*)d_in[0];
    const float* eps_in = (const float*)d_in[1];
    const float* conv_w = (const float*)d_in[2];
    const float* conv_b = (const float*)d_in[3];
    const float* ln1_g  = (const float*)d_in[4];
    const float* ln1_b  = (const float*)d_in[5];
    const float* qkv_w  = (const float*)d_in[6];
    const float* out_w  = (const float*)d_in[7];
    const float* out_b  = (const float*)d_in[8];
    const float* ln2_g  = (const float*)d_in[9];
    const float* ln2_b  = (const float*)d_in[10];
    const float* ff1_w  = (const float*)d_in[11];
    const float* ff1_b  = (const float*)d_in[12];
    const float* ff2_w  = (const float*)d_in[13];
    const float* ff2_b  = (const float*)d_in[14];
    const float* a_mem  = (const float*)d_in[15];
    const float* n_mem  = (const float*)d_in[16];
    const float* mu_w   = (const float*)d_in[17];
    const float* mu_b   = (const float*)d_in[18];
    const float* var_w  = (const float*)d_in[19];
    const float* var_b  = (const float*)d_in[20];

    float* out = (float*)d_out;
    const long OUT_AATT = 3, OUT_NATT = 8195, OUT_ANATT = 16387, OUT_NAATT = 24579;
    const long OUT_COS = 32771, OUT_XOUT = 32772, OUT_VFEAT = 16809988;

    // ---- workspace bump allocator (256B aligned) ----
    char* base = (char*)d_ws;
    size_t woff = 0;
    auto allocB = [&](size_t bytes) -> void* {
        woff = (woff + 255) & ~(size_t)255;
        void* p = base + woff; woff += bytes; return p;
    };
    // f32 buffers
    float* h        = (float*)allocB(16384L * 512 * 4);
    float* scores   = (float*)allocB(64L * 4 * 256 * 256 * 4);
    float* at32_Aam = (float*)allocB(8192L * 64 * 4);
    float* at32_Anm = (float*)allocB(8192L * 64 * 4);
    float* at32_Nam = (float*)allocB(8192L * 64 * 4);
    float* at32_Nnm = (float*)allocB(8192L * 64 * 4);
    float* mu_buf   = (float*)allocB(8192L * 512 * 4);
    float* var_buf  = (float*)allocB(8192L * 512 * 4);
    float* new_buf  = (float*)allocB(8192L * 512 * 4);
    float* Aaug_new = (float*)allocB(8192L * 512 * 4);
    float* ANm      = (float*)allocB(8192L * 512 * 4);
    float* NAm      = (float*)allocB(8192L * 512 * 4);
    float* anchor   = (float*)allocB(32L * 512 * 4);
    float* positive = (float*)allocB(32L * 512 * 4);
    float* negative = (float*)allocB(32L * 512 * 4);
    float* anc_new  = (float*)allocB(32L * 512 * 4);
    float* neg_new  = (float*)allocB(32L * 512 * 4);
    float* trip_b   = (float*)allocB(32 * 4);
    float* cos_b    = (float*)allocB(32 * 4);
    float* dist_b   = (float*)allocB(32 * 4);
    float* kl_part  = (float*)allocB(4096 * 4);
    int*   idxA     = (int*)allocB(1024);
    int*   idxN     = (int*)allocB(1024);
    int*   idxP     = (int*)allocB(1024);
    // f16 buffers
    _Float16* Xcol16   = (_Float16*)allocB(16384L * 3072 * 2);
    _Float16* ln16     = (_Float16*)allocB(16384L * 512 * 2);
    _Float16* qkv16    = (_Float16*)allocB(16384L * 1536 * 2);
    _Float16* probs16  = (_Float16*)allocB(64L * 4 * 256 * 256 * 2);
    _Float16* o16      = (_Float16*)allocB(16384L * 512 * 2);
    _Float16* ff16     = (_Float16*)allocB(16384L * 512 * 2);
    _Float16* h16      = (_Float16*)allocB(16384L * 512 * 2);
    _Float16* at16_Aam = (_Float16*)allocB(8192L * 64 * 2);
    _Float16* at16_Anm = (_Float16*)allocB(8192L * 64 * 2);
    _Float16* at16_Nam = (_Float16*)allocB(8192L * 64 * 2);
    _Float16* at16_Nnm = (_Float16*)allocB(8192L * 64 * 2);
    _Float16* aug_A16  = (_Float16*)allocB(8192L * 512 * 2);   // A_aug
    _Float16* aug_NA16 = (_Float16*)allocB(8192L * 512 * 2);   // N_Aaug
    _Float16* aug_AN16 = (_Float16*)allocB(8192L * 512 * 2);   // A_Naug
    _Float16* aug_N16  = (_Float16*)allocB(8192L * 512 * 2);   // N_aug
    _Float16* convW16  = (_Float16*)allocB(512L * 3072 * 2);
    _Float16* qkvwT16  = (_Float16*)allocB(2L * 1536 * 512 * 2);
    _Float16* outwT16  = (_Float16*)allocB(2L * 512 * 512 * 2);
    _Float16* ff1wT16  = (_Float16*)allocB(2L * 512 * 512 * 2);
    _Float16* ff2wT16  = (_Float16*)allocB(2L * 512 * 512 * 2);
    _Float16* muwT16   = (_Float16*)allocB(512L * 512 * 2);
    _Float16* varwT16  = (_Float16*)allocB(512L * 512 * 2);
    _Float16* memA_a16 = (_Float16*)allocB(64L * 512 * 2);     // padded [64][512]
    _Float16* memA_n16 = (_Float16*)allocB(64L * 512 * 2);
    _Float16* memT_a16 = (_Float16*)allocB(512L * 64 * 2);     // padded [512][64]
    _Float16* memT_n16 = (_Float16*)allocB(512L * 64 * 2);

    const float SCALE  = 0.08838834764831845f;   // 128^-0.5
    const float MSCALE = 0.04419417382415922f;   // 512^-0.5

    // ---- operand prep: f16 conversions / transposes / padding ----
    cvt_f16<<<(512 * 3072 + 255) / 256, 256, 0, stream>>>(conv_w, convW16, 512L * 3072);
    for (int i = 0; i < 2; ++i) {
        cvtT_f16<<<(512 * 1536 + 255) / 256, 256, 0, stream>>>(qkv_w + (long)i * 512 * 1536,
                                                               qkvwT16 + (long)i * 1536 * 512, 512, 1536);
        cvtT_f16<<<(512 * 512 + 255) / 256, 256, 0, stream>>>(out_w + (long)i * 512 * 512,
                                                              outwT16 + (long)i * 512 * 512, 512, 512);
        cvtT_f16<<<(512 * 512 + 255) / 256, 256, 0, stream>>>(ff1_w + (long)i * 512 * 512,
                                                              ff1wT16 + (long)i * 512 * 512, 512, 512);
        cvtT_f16<<<(512 * 512 + 255) / 256, 256, 0, stream>>>(ff2_w + (long)i * 512 * 512,
                                                              ff2wT16 + (long)i * 512 * 512, 512, 512);
    }
    cvtT_f16<<<(512 * 512 + 255) / 256, 256, 0, stream>>>(mu_w, muwT16, 512, 512);
    cvtT_f16<<<(512 * 512 + 255) / 256, 256, 0, stream>>>(var_w, varwT16, 512, 512);
    pad_mem<<<128, 256, 0, stream>>>(a_mem, memA_a16, memT_a16);
    pad_mem<<<128, 256, 0, stream>>>(n_mem, memA_n16, memT_n16);

    // 1) conv1d (im2col + WMMA GEMM, fused bias+relu) -> h (f32)
    im2col_f16<<<(unsigned)((16384L * 3072 + 255) / 256), 256, 0, stream>>>(x, Xcol16, 16384L * 3072);
    launch_gemm(stream, Xcol16, convW16, conv_b, nullptr, h, nullptr, 16384, 512, 3072,
                3072, 1, 3072, 512, 0,0,0,0,0,0, 1, 1, 1.0f, /*relu*/1);

    // 2) transformer layers
    for (int i = 0; i < 2; ++i) {
        layernorm512<<<16384, 256, 0, stream>>>(h, ln1_g + i * 512, ln1_b + i * 512, ln16);
        launch_gemm(stream, ln16, qkvwT16 + (long)i * 1536 * 512, nullptr, nullptr, nullptr, qkv16,
                    16384, 1536, 512, 512, 1, 512, 1536, 0,0,0,0,0,0, 1, 1, 1.0f, 0);
        // scores = SCALE * q @ k^T   (z = b*4 + h)
        launch_gemm(stream, qkv16, qkv16 + 512, nullptr, nullptr, scores, nullptr,
                    256, 256, 128,
                    1536, /*ldbk*/1, /*ldbn*/1536, 256,
                    393216, 128, 393216, 128, 262144, 65536, 4, 256, SCALE, 0);
        softmax256<<<65536, 256, 0, stream>>>(scores, probs16);
        // o = att @ v   (B contiguous in n -> transpose-stage path)
        launch_gemm(stream, probs16, qkv16 + 1024, nullptr, nullptr, nullptr, o16,
                    256, 128, 256,
                    256, /*ldbk*/1536, /*ldbn*/1, 512,
                    262144, 65536, 393216, 128, 131072, 128, 4, 256, 1.0f, 0);
        // h += o @ out_w + out_b
        launch_gemm(stream, o16, outwT16 + (long)i * 512 * 512, out_b + i * 512, h, h, nullptr,
                    16384, 512, 512, 512, 1, 512, 512, 0,0,0,0,0,0, 1, 1, 1.0f, 0);
        // FF block
        layernorm512<<<16384, 256, 0, stream>>>(h, ln2_g + i * 512, ln2_b + i * 512, ln16);
        launch_gemm(stream, ln16, ff1wT16 + (long)i * 512 * 512, ff1_b + i * 512, nullptr, nullptr, ff16,
                    16384, 512, 512, 512, 1, 512, 512, 0,0,0,0,0,0, 1, 1, 1.0f, /*gelu*/2);
        launch_gemm(stream, ff16, ff2wT16 + (long)i * 512 * 512, ff2_b + i * 512, h, h, nullptr,
                    16384, 512, 512, 512, 1, 512, 512, 0,0,0,0,0,0, 1, 1, 1.0f, 0);
    }

    cvt_f16<<<(unsigned)((16384L * 512 + 255) / 256), 256, 0, stream>>>(h, h16, 16384L * 512);
    const float*    N_x   = h;
    const float*    A_x   = h + 8192L * 512;
    const _Float16* N_x16 = h16;
    const _Float16* A_x16 = h16 + 8192L * 512;

    // 3) memory attention: sigmoid(x @ mem^T / sqrt(D)); N padded 60->64
    launch_gemm(stream, A_x16, memA_a16, nullptr, nullptr, at32_Aam, at16_Aam, 8192, 64, 512,
                512, 1, 512, 64, 0,0,0,0,0,0, 1, 1, MSCALE, 3);
    launch_gemm(stream, A_x16, memA_n16, nullptr, nullptr, at32_Anm, at16_Anm, 8192, 64, 512,
                512, 1, 512, 64, 0,0,0,0,0,0, 1, 1, MSCALE, 3);
    launch_gemm(stream, N_x16, memA_a16, nullptr, nullptr, at32_Nam, at16_Nam, 8192, 64, 512,
                512, 1, 512, 64, 0,0,0,0,0,0, 1, 1, MSCALE, 3);
    launch_gemm(stream, N_x16, memA_n16, nullptr, nullptr, at32_Nnm, at16_Nnm, 8192, 64, 512,
                512, 1, 512, 64, 0,0,0,0,0,0, 1, 1, MSCALE, 3);

    // 4) temporal attentions (top-4 mean over real 60 cols) -> d_out
    top4mean<<<32, 256, 0, stream>>>(at32_Aam, out + OUT_AATT,  8192, 60, 64);  // A_att
    top4mean<<<32, 256, 0, stream>>>(at32_Anm, out + OUT_NAATT, 8192, 60, 64);  // N_Aatt
    top4mean<<<32, 256, 0, stream>>>(at32_Nam, out + OUT_ANATT, 8192, 60, 64);  // A_Natt
    top4mean<<<32, 256, 0, stream>>>(at32_Nnm, out + OUT_NATT,  8192, 60, 64);  // N_att

    // 5) augmentations: att @ mem (K padded to 64; pad rows of memT are zero)
    launch_gemm(stream, at16_Aam, memT_a16, nullptr, nullptr, nullptr, aug_A16,  8192, 512, 64,
                64, 1, 64, 512, 0,0,0,0,0,0, 1, 1, 1.0f, 0);
    launch_gemm(stream, at16_Anm, memT_n16, nullptr, nullptr, nullptr, aug_NA16, 8192, 512, 64,
                64, 1, 64, 512, 0,0,0,0,0,0, 1, 1, 1.0f, 0);
    launch_gemm(stream, at16_Nam, memT_a16, nullptr, nullptr, nullptr, aug_AN16, 8192, 512, 64,
                64, 1, 64, 512, 0,0,0,0,0,0, 1, 1, 1.0f, 0);
    launch_gemm(stream, at16_Nnm, memT_n16, nullptr, nullptr, nullptr, aug_N16,  8192, 512, 64,
                64, 1, 64, 512, 0,0,0,0,0,0, 1, 1, 1.0f, 0);

    // 6) top-17 over T + gather-means
    const int K17 = 17;
    topk_idx<<<1, 32, 0, stream>>>(out + OUT_AATT,  idxA, 32, 256, K17);
    topk_idx<<<1, 32, 0, stream>>>(out + OUT_NATT,  idxN, 32, 256, K17);
    topk_idx<<<1, 32, 0, stream>>>(out + OUT_NAATT, idxP, 32, 256, K17);
    gather_mean<<<dim3(32, 2), 256, 0, stream>>>(A_x, idxA, negative, 256, 512, K17);
    gather_mean<<<dim3(32, 2), 256, 0, stream>>>(N_x, idxN, anchor,   256, 512, K17);
    gather_mean<<<dim3(32, 2), 256, 0, stream>>>(A_x, idxP, positive, 256, 512, K17);

    // 7) mu/var projections
    launch_gemm(stream, aug_N16,  muwT16,  mu_b,  nullptr, mu_buf,   nullptr, 8192, 512, 512,
                512, 1, 512, 512, 0,0,0,0,0,0, 1, 1, 1.0f, 0);
    launch_gemm(stream, aug_N16,  varwT16, var_b, nullptr, var_buf,  nullptr, 8192, 512, 512,
                512, 1, 512, 512, 0,0,0,0,0,0, 1, 1, 1.0f, 0);
    launch_gemm(stream, aug_A16,  muwT16,  mu_b,  nullptr, Aaug_new, nullptr, 8192, 512, 512,
                512, 1, 512, 512, 0,0,0,0,0,0, 1, 1, 1.0f, 0);
    launch_gemm(stream, aug_AN16, muwT16,  mu_b,  nullptr, ANm,      nullptr, 8192, 512, 512,
                512, 1, 512, 512, 0,0,0,0,0,0, 1, 1, 1.0f, 0);
    launch_gemm(stream, aug_NA16, muwT16,  mu_b,  nullptr, NAm,      nullptr, 8192, 512, 512,
                512, 1, 512, 512, 0,0,0,0,0,0, 1, 1, 1.0f, 0);

    // 8) reparameterize + gathers
    vae_combine<<<16384, 256, 0, stream>>>(mu_buf, var_buf, eps_in, new_buf, 8192L * 512);
    gather_mean<<<dim3(32, 2), 256, 0, stream>>>(new_buf,  idxN, anc_new, 256, 512, K17);
    gather_mean<<<dim3(32, 2), 256, 0, stream>>>(Aaug_new, idxA, neg_new, 256, 512, K17);

    // 9) losses
    kl_partial<<<4096, 256, 0, stream>>>(mu_buf, var_buf, kl_part, 8192L * 512);
    tripcos_kernel<<<32, 256, 0, stream>>>(anchor, positive, negative, trip_b, cos_b);
    dist_kernel<<<32, 256, 0, stream>>>(anc_new, neg_new, dist_b);
    finalize_scalars<<<1, 256, 0, stream>>>(kl_part, 4096, trip_b, cos_b, dist_b, out, OUT_COS);

    // 10) x_out + v_feat
    assemble_out<<<32768, 256, 0, stream>>>(h, new_buf, ANm, Aaug_new, NAm,
                                            out + OUT_XOUT, out + OUT_VFEAT);
}
